// SimpleRetention_25872882991701
// MI455X (gfx1250) — matrix operand order repaired and checked by reading, not compiled, not run on hardware
//
#include <hip/hip_runtime.h>
#include <hip/hip_bf16.h>

// ---------------------------------------------------------------------------
// SimpleRetention on MI455X (gfx1250), wave32 + WMMA.
//   Stage 1: Q/K/V projections (X@W) via v_wmma_f32_16x16x32_f16, xPos epilogue
//            (reciprocal-mul + hoisted per-column transcendentals), f16 out.
//   Stage 2: flash-style causal retention. Per 32-key tile:
//            waves 0-3 each compute one DISTINCT 16x16 S = Q K^T fragment
//            (no duplicated WMMA work), apply decay gamma^(q-k) via exp2
//            (underflows to 0 for large deltas -> no renormalization), stash
//            S as f16 in LDS; all 8 waves then do O += S @ V.
//            Q/K tiles staged with global_load_async_to_lds_b128 (ASYNCcnt
//            path, confirmed present on this toolchain), drained with
//            s_wait_asynccnt before the workgroup barrier.
// ---------------------------------------------------------------------------

typedef __attribute__((ext_vector_type(16))) _Float16 v16h;
typedef __attribute__((ext_vector_type(8)))  float    v8f;
typedef __attribute__((ext_vector_type(4)))  int      v4i;

#define HEAD_DIM   256
#define HIDDEN     1024
#define SEQ        4096
#define BATCH      4
// log2(0.96875)
#define LOG2_GAMMA  (-0.04580368961312478f)
// log2(10000)/128
#define L2_10K_O128 (0.10381025296523006f)

#define AS1 __attribute__((address_space(1)))
#define AS3 __attribute__((address_space(3)))

union F16Frag { v16h v; uint4 q[2]; };

static __device__ inline v8f wmma_f16(v16h a, v16h b, v8f c) {
  // 8 args: (neg_a, A, neg_b, B, c_mod, C, reuse_a, reuse_b)
  return __builtin_amdgcn_wmma_f32_16x16x32_f16(false, a, false, b, (short)0, c,
                                                false, false);
}

// 16-byte global -> LDS stage; async (ASYNCcnt) path when available.
static __device__ inline void stage16(const _Float16* __restrict__ g,
                                      _Float16* l) {
#if __has_builtin(__builtin_amdgcn_global_load_async_to_lds_b128)
  // signature: (v4i AS1* src, v4i AS3* dst, imm int offset, imm int cpol)
  __builtin_amdgcn_global_load_async_to_lds_b128(
      (AS1 v4i*)(AS1 void*)g, (AS3 v4i*)(AS3 void*)l, 0, 0);
#else
  *(uint4*)l = *(const uint4*)g;
#endif
}

static __device__ inline void stage_wait() {
#if __has_builtin(__builtin_amdgcn_global_load_async_to_lds_b128)
#if __has_builtin(__builtin_amdgcn_s_wait_asynccnt)
  __builtin_amdgcn_s_wait_asynccnt(0);
#else
  asm volatile("s_wait_asynccnt 0" ::: "memory");
#endif
#endif
}

// ---------------------------------------------------------------------------
// Kernel 1: fused projection + xPos. block = 256 threads (8 waves).
// Block computes a 32-row x 256-col tile of X@W for one of Q/K/V (blockIdx.y).
// ---------------------------------------------------------------------------
__global__ __launch_bounds__(256) void retn_proj_xpos(
    const float* __restrict__ X,
    const float* __restrict__ Wq,
    const float* __restrict__ Wk,
    const float* __restrict__ Wv,
    _Float16* __restrict__ Qh,
    _Float16* __restrict__ Kh,
    _Float16* __restrict__ Vh)
{
  __shared__ __align__(16) _Float16 Xs[32][40];    // [row][k], padded
  __shared__ __align__(16) _Float16 Wst[256][40];  // transposed: [col][k]

  const int mode = blockIdx.y;  // 0=Q, 1=K(downscale), 2=V(no xpos)
  const float*  W   = (mode == 0) ? Wq : (mode == 1) ? Wk : Wv;
  _Float16*     Out = (mode == 0) ? Qh : (mode == 1) ? Kh : Vh;

  const int t      = threadIdx.x;
  const int lane   = t & 31;
  const int wave   = t >> 5;
  const int hi     = lane >> 4;
  const int ln     = lane & 15;
  const int rowSub = wave >> 2;   // 0..1
  const int colQ   = wave & 3;    // 0..3 (64 cols each)
  const int rowBase = blockIdx.x * 32;

  const int r8 = t >> 3;          // cooperative row 0..31
  const int c8 = (t & 7) * 4;     // cooperative col group

  const v8f vzero = {0.f, 0.f, 0.f, 0.f, 0.f, 0.f, 0.f, 0.f};
  v8f acc[4];
#pragma unroll
  for (int f = 0; f < 4; ++f) acc[f] = vzero;

  for (int k0 = 0; k0 < HIDDEN; k0 += 32) {
    {
      const float4 x4 =
          *(const float4*)&X[(size_t)(rowBase + r8) * HIDDEN + k0 + c8];
      Xs[r8][c8 + 0] = (_Float16)x4.x;
      Xs[r8][c8 + 1] = (_Float16)x4.y;
      Xs[r8][c8 + 2] = (_Float16)x4.z;
      Xs[r8][c8 + 3] = (_Float16)x4.w;
    }
#pragma unroll
    for (int g = 0; g < 8; ++g) {
      const int c = c8 + 32 * g;
      const float4 w4 = *(const float4*)&W[(size_t)(k0 + r8) * HEAD_DIM + c];
      Wst[c + 0][r8] = (_Float16)w4.x;
      Wst[c + 1][r8] = (_Float16)w4.y;
      Wst[c + 2][r8] = (_Float16)w4.z;
      Wst[c + 3][r8] = (_Float16)w4.w;
    }
    __syncthreads();

    F16Frag a;  // A frag: row = ln, K halves [hi*8..hi*8+7], [16+hi*8..]
    a.q[0] = *(const uint4*)&Xs[rowSub * 16 + ln][hi * 8];
    a.q[1] = *(const uint4*)&Xs[rowSub * 16 + ln][16 + hi * 8];

#pragma unroll
    for (int f = 0; f < 4; ++f) {
      F16Frag b;  // B frag: col = ln, K halves [hi*8..hi*8+7], [16+hi*8..]
      const int col = colQ * 64 + f * 16 + ln;
      b.q[0] = *(const uint4*)&Wst[col][hi * 8];
      b.q[1] = *(const uint4*)&Wst[col][16 + hi * 8];
      acc[f] = wmma_f16(a.v, b.v, acc[f]);
    }
    __syncthreads();
  }

  // xPos epilogue. Pair partner (d^1) lives in the adjacent lane -> shfl_xor.
  // Column-dependent factors hoisted out of the row loop; division replaced
  // by reciprocal multiply (avoids v_div_scale/v_div_fixup sequences).
#pragma unroll
  for (int f = 0; f < 4; ++f) {
    const int d = colQ * 64 + f * 16 + ln;
    float log2sb = 0.f, invf = 0.f, esgn = 0.f;
    if (mode != 2) {
      log2sb = __log2f(((float)(d & ~1) + 102.4f) * (1.0f / 358.4f));
      invf   = exp2f(-L2_10K_O128 * (float)(d >> 1));
      esgn   = (mode == 1) ? -(1.0f / 512.0f) : (1.0f / 512.0f);
    }
#pragma unroll
    for (int j = 0; j < 8; ++j) {
      const int row = rowBase + rowSub * 16 + j + hi * 8;
      const int pos = row & (SEQ - 1);
      float x  = acc[f][j];
      float xp = __shfl_xor(x, 1, 32);
      float outv;
      if (mode == 2) {
        outv = x;  // V: plain projection
      } else {
        const float scl = exp2f((float)pos * esgn * log2sb);
        float s, c;
        __sincosf((float)pos * invf, &s, &c);
        const float rot = (d & 1) ? xp : -xp;
        outv = (x * c + rot * s) * scl;
      }
      Out[(size_t)row * HEAD_DIM + d] = (_Float16)outv;
    }
  }
}

// ---------------------------------------------------------------------------
// Kernel 2: causal retention attention. block = 256 threads (8 waves).
// Block: 32 q-rows x 256 out cols for one batch. Wave w: qSub = w&1,
// colQ = w>>1. Waves 0-3 own the four distinct S fragments (no duplicated
// WMMA); S is shared through LDS; all waves do the S@V stage.
// ---------------------------------------------------------------------------
__global__ __launch_bounds__(256) void retn_attn(
    const _Float16* __restrict__ Qh,
    const _Float16* __restrict__ Kh,
    const _Float16* __restrict__ Vh,
    float* __restrict__ Out)
{
  __shared__ __align__(16) _Float16 KsBuf[32 * 264];  // K tile [key][dim]
  __shared__ __align__(16) _Float16 Vst[256][40];     // V transposed [col][key]
  __shared__ __align__(16) _Float16 sS[32][40];       // decayed S [q][key]
  _Float16 (*Ks)[264] = (_Float16(*)[264])KsBuf;
  _Float16 (*Qs)[264] = (_Float16(*)[264])KsBuf;      // overlap: startup only

  const int t    = threadIdx.x;
  const int lane = t & 31;
  const int wave = t >> 5;
  const int hi   = lane >> 4;
  const int ln   = lane & 15;
  const int qSub = wave & 1;   // q-subtile for this wave (S and O stages)
  const int colQ = wave >> 1;  // 64-col quarter for O stage
  const int qbase = blockIdx.x * 32;
  const int b     = blockIdx.y;
  const size_t bOff = (size_t)b * SEQ * HEAD_DIM;

  const int r8  = t >> 3;        // 0..31
  const int c32 = (t & 7) * 32;  // 8 threads per row, 32 halves each

  // stage Q tile (into the Ks region; Q is register-resident afterwards)
#pragma unroll
  for (int i = 0; i < 4; ++i)
    stage16(&Qh[bOff + (size_t)(qbase + r8) * HEAD_DIM + c32 + i * 8],
            &Qs[r8][c32 + i * 8]);
  stage_wait();
  __syncthreads();

  F16Frag qA[8];
  if (wave < 4) {  // only S-waves need Q fragments
#pragma unroll
    for (int kk = 0; kk < 8; ++kk) {
      qA[kk].q[0] = *(const uint4*)&Qs[qSub * 16 + ln][kk * 32 + hi * 8];
      qA[kk].q[1] = *(const uint4*)&Qs[qSub * 16 + ln][kk * 32 + 16 + hi * 8];
    }
  }
  __syncthreads();  // release Qs region; it becomes the K tile

  const v8f vzero = {0.f, 0.f, 0.f, 0.f, 0.f, 0.f, 0.f, 0.f};
  v8f oacc[4];
#pragma unroll
  for (int f = 0; f < 4; ++f) oacc[f] = vzero;

  const int nT = qbase / 32 + 1;  // key tiles up to (incl.) the diagonal
  for (int tile = 0; tile < nT; ++tile) {
    const int m0 = tile * 32;

    // stage K tile (async path) and V tile transposed (manual)
#pragma unroll
    for (int i = 0; i < 4; ++i)
      stage16(&Kh[bOff + (size_t)(m0 + r8) * HEAD_DIM + c32 + i * 8],
              &Ks[r8][c32 + i * 8]);
#pragma unroll
    for (int i = 0; i < 4; ++i) {
      uint4 v = *(const uint4*)&Vh[bOff + (size_t)(m0 + r8) * HEAD_DIM +
                                   c32 + i * 8];
      const _Float16* h8 = (const _Float16*)&v;
#pragma unroll
      for (int e = 0; e < 8; ++e) Vst[c32 + i * 8 + e][r8] = h8[e];
    }
    if (tile + 1 < nT) {  // global_prefetch_b8 for the next tile
      __builtin_prefetch(&Kh[bOff + (size_t)(m0 + 32 + r8) * HEAD_DIM + c32], 0, 1);
      __builtin_prefetch(&Vh[bOff + (size_t)(m0 + 32 + r8) * HEAD_DIM + c32], 0, 1);
    }
    stage_wait();
    __syncthreads();

    // S stage: waves 0-3, each one distinct 16q x 16key fragment.
    if (wave < 4) {
      const int sk = wave >> 1;  // key 16-chunk; q chunk is qSub
      v8f sacc = vzero;
#pragma unroll
      for (int kk = 0; kk < 8; ++kk) {
        F16Frag bF;  // B[k=dim][n=key] = K[key][dim]
        bF.q[0] = *(const uint4*)&Ks[sk * 16 + ln][kk * 32 + hi * 8];
        bF.q[1] = *(const uint4*)&Ks[sk * 16 + ln][kk * 32 + 16 + hi * 8];
        sacc = wmma_f16(qA[kk].v, bF.v, sacc);
      }
#pragma unroll
      for (int j = 0; j < 8; ++j) {
        const int q   = qbase + qSub * 16 + j + hi * 8;
        const int key = m0 + sk * 16 + ln;
        const int dlt = q - key;
        float val = (dlt >= 0) ? sacc[j] * exp2f((float)dlt * LOG2_GAMMA) : 0.f;
        sS[qSub * 16 + j + hi * 8][sk * 16 + ln] = (_Float16)val;
      }
    }
    __syncthreads();

    // O stage: all 8 waves, O += S @ V.
    F16Frag sA;  // A frag of S: row = ln (q), 32 keys
    sA.q[0] = *(const uint4*)&sS[qSub * 16 + ln][hi * 8];
    sA.q[1] = *(const uint4*)&sS[qSub * 16 + ln][16 + hi * 8];
#pragma unroll
    for (int f = 0; f < 4; ++f) {
      F16Frag bF;  // B[k=key][n=col] = Vst[col][key]
      const int col = colQ * 64 + f * 16 + ln;
      bF.q[0] = *(const uint4*)&Vst[col][hi * 8];
      bF.q[1] = *(const uint4*)&Vst[col][16 + hi * 8];
      oacc[f] = wmma_f16(sA.v, bF.v, oacc[f]);
    }
    __syncthreads();
  }

  // write O (f32)
#pragma unroll
  for (int f = 0; f < 4; ++f) {
#pragma unroll
    for (int j = 0; j < 8; ++j) {
      const int q = qbase + qSub * 16 + j + hi * 8;
      const int c = colQ * 64 + f * 16 + ln;
      Out[bOff + (size_t)q * HEAD_DIM + c] = oacc[f][j];
    }
  }
}

// ---------------------------------------------------------------------------
extern "C" void kernel_launch(void* const* d_in, const int* in_sizes, int n_in,
                              void* d_out, int out_size, void* d_ws, size_t ws_size,
                              hipStream_t stream) {
  const float* X  = (const float*)d_in[0];  // (4, 4096, 1024)
  const float* Wq = (const float*)d_in[1];  // (1024, 256)
  const float* Wk = (const float*)d_in[2];
  const float* Wv = (const float*)d_in[3];
  float* out = (float*)d_out;               // (4, 4096, 256)

  const size_t nQ = (size_t)BATCH * SEQ * HEAD_DIM;
  _Float16* Qh = (_Float16*)d_ws;           // 25.2 MB f16 workspace
  _Float16* Kh = Qh + nQ;
  _Float16* Vh = Kh + nQ;

  dim3 g1((BATCH * SEQ) / 32, 3);  // 512 row-tiles x {Q,K,V}
  retn_proj_xpos<<<g1, 256, 0, stream>>>(X, Wq, Wk, Wv, Qh, Kh, Vh);

  dim3 g2(SEQ / 32, BATCH);        // 128 q-tiles x 4 batches
  retn_attn<<<g2, 256, 0, stream>>>(Qh, Kh, Vh, out);
}
